// FullARModel_9234179686510
// MI455X (gfx1250) — matrix operand order repaired and checked by reading, not compiled, not run on hardware
//
#include <hip/hip_runtime.h>
#include <math.h>

// ---------------------------------------------------------------------------
// LSTM LM on MI455X (gfx1250): bf16 WMMA (v_wmma_f32_16x16x32_bf16) for all
// matmuls, f32 accumulation. soft_W (bf16, 20.5MB) is L2-resident on CDNA5
// (192MB L2), so the 60 decode-phase 128x20000x512 matmuls stream from L2.
// Each wave computes a 16x32 output tile: one A fragment feeds two WMMAs
// (doubles FLOP/byte of fragment traffic vs a 16x16 tile per wave).
// ---------------------------------------------------------------------------

typedef __attribute__((ext_vector_type(16))) __bf16 v16bf;
typedef __attribute__((ext_vector_type(8)))  float  v8f;

#define VOCAB 20000
#define UNITS 512
#define EMB   64
#define BSZ   128
#define TLEN  64
#define NDEC  59

// float -> bf16, round-to-nearest-even
__device__ __forceinline__ unsigned short f2bf(float f) {
  union { float f; unsigned u; } x; x.f = f;
  unsigned u = x.u;
  u += 0x7FFFu + ((u >> 16) & 1u);
  return (unsigned short)(u >> 16);
}

// CDNA5 WMMA 16-bit A-matrix (16x32 MxK) element K-index for (lane, elem j)
// lanes 0-15: v0..v3 -> K 0..7, v4..v7 -> K 16..23 ; lanes 16-31: +8
__device__ __forceinline__ int kofA(int lane, int j) {
  int v = j >> 1, h = j & 1;
  return ((v & 4) ? 16 : 0) + ((lane >> 4) << 3) + ((v & 3) << 1) + h;
}
// CDNA5 WMMA 16-bit B-matrix (32x16 KxN): same K order as A (lanes 0-15: K 0..7,16..23; lanes 16-31: +8)
__device__ __forceinline__ int kofB(int lane, int j) {
  int v = j >> 1, h = j & 1;
  return ((v & 4) ? 16 : 0) + ((lane >> 4) << 3) + ((v & 3) << 1) + h;
}

__device__ __forceinline__ float sigmoidf(float x) {
  return 1.0f / (1.0f + __expf(-x));
}

// ---- pack row-major f32 weight [K x N] into B-fragment-tiled bf16 ----------
__global__ void pack_B(const float* __restrict__ W, int N, int ktiles,
                       unsigned short* __restrict__ out, int total) {
  int idx = blockIdx.x * blockDim.x + threadIdx.x;
  if (idx >= total) return;
  int j    = idx & 15;
  int lane = (idx >> 4) & 31;
  int t    = idx >> 9;          // nt*ktiles + kt
  int kt   = t % ktiles;
  int nt   = t / ktiles;
  int k = kt * 32 + kofB(lane, j);
  int n = nt * 16 + (lane & 15);
  out[idx] = f2bf(W[(size_t)k * N + n]);
}

// ---- pack concatenated [lstm_K ; lstm_R] (576 x 2048) into B tiles ---------
__global__ void pack_B_lstm(const float* __restrict__ Kw,
                            const float* __restrict__ Rw,
                            unsigned short* __restrict__ out) {
  int idx = blockIdx.x * blockDim.x + threadIdx.x;
  if (idx >= 128 * 18 * 512) return;
  int j    = idx & 15;
  int lane = (idx >> 4) & 31;
  int t    = idx >> 9;
  int kt   = t % 18;
  int nt   = t / 18;
  int k = kt * 32 + kofB(lane, j);
  int n = nt * 16 + (lane & 15);
  float v = (k < EMB) ? Kw[(size_t)k * (4 * UNITS) + n]
                      : Rw[(size_t)(k - EMB) * (4 * UNITS) + n];
  out[idx] = f2bf(v);
}

// ---- pack LSTM input [emb(tok) | h] (128 x 576) into A tiles ---------------
__global__ void pack_A_lstm(const float* __restrict__ emb,
                            const int* __restrict__ toks, int tstride, int toff,
                            const float* __restrict__ h,
                            unsigned short* __restrict__ out) {
  int idx = blockIdx.x * blockDim.x + threadIdx.x;
  if (idx >= 8 * 18 * 512) return;
  int j    = idx & 15;
  int lane = (idx >> 4) & 31;
  int t    = idx >> 9;
  int kt   = t % 18;
  int mt   = t / 18;
  int m = mt * 16 + (lane & 15);
  int k = kt * 32 + kofA(lane, j);
  float v;
  if (k < EMB) {
    int tok = toks[m * tstride + toff];
    v = emb[(size_t)tok * EMB + k];
  } else {
    v = h[m * UNITS + (k - EMB)];
  }
  out[idx] = f2bf(v);
}

// ---- pack 128x512 f32 activation into A tiles ------------------------------
__global__ void pack_A512(const float* __restrict__ src,
                          unsigned short* __restrict__ out) {
  int idx = blockIdx.x * blockDim.x + threadIdx.x;
  if (idx >= 8 * 16 * 512) return;
  int j    = idx & 15;
  int lane = (idx >> 4) & 31;
  int t    = idx >> 9;
  int kt   = t % 16;
  int mt   = t / 16;
  int m = mt * 16 + (lane & 15);
  int k = kt * 32 + kofA(lane, j);
  out[idx] = f2bf(src[m * UNITS + k]);
}

// ---- generic bf16 WMMA matmul: C[128 x ldC] = A @ B + bias (opt relu) ------
// one 16x32 tile per wave (2 accumulators, shared A fragment);
// 8 waves/block share the same N-tile pair so B fragments hit in-cache.
__global__ void mm_wmma(const unsigned short* __restrict__ A,
                        const unsigned short* __restrict__ Bm,
                        const float* __restrict__ bias,
                        float* __restrict__ C, int ldC, int ktiles, int relu) {
  int lane = threadIdx.x & 31;
  int mt   = threadIdx.x >> 5;   // 0..7  (M = 128 -> 8 tiles)
  int nt0  = blockIdx.x * 2;
  const unsigned short* ap = A  + (((size_t)mt  * ktiles) * 32 + lane) * 16;
  const unsigned short* b0 = Bm + (((size_t)nt0 * ktiles) * 32 + lane) * 16;
  const unsigned short* b1 = b0 + (size_t)ktiles * 512;
  v8f acc0 = {};
  v8f acc1 = {};
  for (int kt = 0; kt < ktiles; ++kt) {
    v16bf a   = *(const v16bf*)ap;
    v16bf bb0 = *(const v16bf*)b0;
    v16bf bb1 = *(const v16bf*)b1;
    // speculative prefetch of the next fragments; dropped if past the end
    __builtin_prefetch(b0 + 512, 0, 1);
    __builtin_prefetch(b1 + 512, 0, 1);
    acc0 = __builtin_amdgcn_wmma_f32_16x16x32_bf16(
        false, a, false, bb0, (short)0, acc0, false, false);
    acc1 = __builtin_amdgcn_wmma_f32_16x16x32_bf16(
        false, a, false, bb1, (short)0, acc1, false, false);
    ap += 512;
    b0 += 512;
    b1 += 512;
  }
  // C/D layout: VGPR r -> M = r + (lane>=16)*8 ; N = lane&15
  int mbase = mt * 16 + ((lane >> 4) << 3);
  #pragma unroll
  for (int half = 0; half < 2; ++half) {
    int n = (nt0 + half) * 16 + (lane & 15);
    float bv = bias[n];
    v8f accv = half ? acc1 : acc0;
    #pragma unroll
    for (int r = 0; r < 8; ++r) {
      float v = accv[r] + bv;
      if (relu) v = fmaxf(v, 0.0f);
      C[(size_t)(mbase + r) * ldC + n] = v;
    }
  }
}

// ---- LSTM gates + state update (optional warmup mask) ----------------------
__global__ void lstm_gates(const float* __restrict__ z,
                           float* __restrict__ h, float* __restrict__ c,
                           const int* __restrict__ inputs, int t) {
  int idx = blockIdx.x * blockDim.x + threadIdx.x;
  if (idx >= BSZ * UNITS) return;
  int b = idx >> 9;
  int n = idx & (UNITS - 1);
  const float* zr = z + (size_t)b * (4 * UNITS);
  float gi = zr[n];
  float gf = zr[UNITS + n];
  float gg = zr[2 * UNITS + n];
  float go = zr[3 * UNITS + n];
  float cn = sigmoidf(gf) * c[idx] + sigmoidf(gi) * tanhf(gg);
  float hn = sigmoidf(go) * tanhf(cn);
  bool upd = true;
  if (inputs) upd = (inputs[b * TLEN + t] != 0);
  if (upd) { c[idx] = cn; h[idx] = hn; }
}

// ---- per-row: max + argmax (softmax-monotone => argmax of pred) + sum(exp) -
__global__ void row_stats(const float* __restrict__ logits,
                          float* __restrict__ rmax, float* __restrict__ rsum,
                          int* __restrict__ tok) {
  __shared__ float smax[256];
  __shared__ int   sidx[256];
  __shared__ float ssum[256];
  int row = blockIdx.x, tid = threadIdx.x;
  const float* lr = logits + (size_t)row * VOCAB;
  float best = -3.4e38f; int bidx = 0;
  for (int n = tid; n < VOCAB; n += 256) {
    float v = lr[n];
    if (v > best) { best = v; bidx = n; }
  }
  smax[tid] = best; sidx[tid] = bidx;
  __syncthreads();
  for (int s = 128; s > 0; s >>= 1) {
    if (tid < s) {
      if (smax[tid + s] > smax[tid] ||
          (smax[tid + s] == smax[tid] && sidx[tid + s] < sidx[tid])) {
        smax[tid] = smax[tid + s];
        sidx[tid] = sidx[tid + s];
      }
    }
    __syncthreads();
  }
  float m = smax[0];
  float sum = 0.0f;
  for (int n = tid; n < VOCAB; n += 256) sum += __expf(lr[n] - m);
  ssum[tid] = sum;
  __syncthreads();
  for (int s = 128; s > 0; s >>= 1) {
    if (tid < s) ssum[tid] += ssum[tid + s];
    __syncthreads();
  }
  if (tid == 0) { rmax[row] = m; rsum[row] = ssum[0]; tok[row] = sidx[0]; }
}

// ---- pred = softmax(logits); acc = pred (mode 0) or acc += pred (mode 1) ---
__global__ void softmax_accum(const float* __restrict__ logits,
                              const float* __restrict__ rmax,
                              const float* __restrict__ rsum,
                              float* __restrict__ acc, int mode) {
  int idx = blockIdx.x * blockDim.x + threadIdx.x;
  if (idx >= BSZ * VOCAB) return;
  int row = idx / VOCAB;
  float p = __expf(logits[idx] - rmax[row]) / rsum[row];
  if (mode) acc[idx] += p; else acc[idx] = p;
}

__global__ void zero_f32(float* __restrict__ p, int n) {
  int idx = blockIdx.x * blockDim.x + threadIdx.x;
  if (idx < n) p[idx] = 0.0f;
}

// ---------------------------------------------------------------------------
extern "C" void kernel_launch(void* const* d_in, const int* in_sizes, int n_in,
                              void* d_out, int out_size, void* d_ws, size_t ws_size,
                              hipStream_t stream) {
  (void)in_sizes; (void)n_in; (void)out_size; (void)ws_size;
  const int*   inputs = (const int*)  d_in[0];
  const float* emb    = (const float*)d_in[1];
  const float* lstmK  = (const float*)d_in[2];
  const float* lstmR  = (const float*)d_in[3];
  const float* lstmb  = (const float*)d_in[4];
  const float* denseW = (const float*)d_in[5];
  const float* denseb = (const float*)d_in[6];
  const float* softW  = (const float*)d_in[7];
  const float* softb  = (const float*)d_in[8];
  float* acc = (float*)d_out;

  char* ws = (char*)d_ws;
  size_t off = 0;
  auto alloc = [&](size_t bytes) -> char* {
    char* p = ws + off;
    off = (off + bytes + 255) & ~(size_t)255;
    return p;
  };
  unsigned short* softW_t  = (unsigned short*)alloc((size_t)1250 * 16 * 512 * 2); // 20.5 MB
  unsigned short* lstmW_t  = (unsigned short*)alloc((size_t)128  * 18 * 512 * 2);
  unsigned short* denseW_t = (unsigned short*)alloc((size_t)32   * 16 * 512 * 2);
  float* h      = (float*)alloc((size_t)BSZ * UNITS * 4);
  float* c      = (float*)alloc((size_t)BSZ * UNITS * 4);   // contiguous after h
  float* z      = (float*)alloc((size_t)BSZ * 4 * UNITS * 4);
  float* dact   = (float*)alloc((size_t)BSZ * UNITS * 4);
  float* logits = (float*)alloc((size_t)BSZ * VOCAB * 4);
  unsigned short* A_lstm = (unsigned short*)alloc((size_t)8 * 18 * 512 * 2);
  unsigned short* A_act  = (unsigned short*)alloc((size_t)8 * 16 * 512 * 2);
  float* rmax = (float*)alloc(BSZ * 4);
  float* rsum = (float*)alloc(BSZ * 4);
  int*   toks = (int*)  alloc(BSZ * 4);

  // one-time weight repacks (f32 -> WMMA-fragment-tiled bf16)
  pack_B<<<40000, 256, 0, stream>>>(softW,  VOCAB, 16, softW_t, 1250 * 16 * 512);
  pack_B<<<1024,  256, 0, stream>>>(denseW, UNITS, 16, denseW_t,  32 * 16 * 512);
  pack_B_lstm<<<4608, 256, 0, stream>>>(lstmK, lstmR, lstmW_t);
  zero_f32<<<512, 256, 0, stream>>>(h, 2 * BSZ * UNITS);   // zeros h and c

  // ---- warm phase: 64 masked LSTM steps over the input sequence ----
  for (int t = 0; t < TLEN; ++t) {
    pack_A_lstm<<<288, 256, 0, stream>>>(emb, inputs, TLEN, t, h, A_lstm);
    mm_wmma<<<64, 256, 0, stream>>>(A_lstm, lstmW_t, lstmb, z, 4 * UNITS, 18, 0);
    lstm_gates<<<256, 256, 0, stream>>>(z, h, c, inputs, t);
  }

  // ---- deep(h) once, then 59 autoregressive decode steps ----
  for (int s = 0; s <= NDEC; ++s) {
    if (s > 0) {
      pack_A_lstm<<<288, 256, 0, stream>>>(emb, toks, 1, 0, h, A_lstm);
      mm_wmma<<<64, 256, 0, stream>>>(A_lstm, lstmW_t, lstmb, z, 4 * UNITS, 18, 0);
      lstm_gates<<<256, 256, 0, stream>>>(z, h, c, (const int*)nullptr, 0);
    }
    pack_A512<<<256, 256, 0, stream>>>(h, A_act);
    mm_wmma<<<16, 256, 0, stream>>>(A_act, denseW_t, denseb, dact, UNITS, 16, 1);
    pack_A512<<<256, 256, 0, stream>>>(dact, A_act);
    mm_wmma<<<625, 256, 0, stream>>>(A_act, softW_t, softb, logits, VOCAB, 16, 0);
    row_stats<<<BSZ, 256, 0, stream>>>(logits, rmax, rsum, toks);
    softmax_accum<<<10000, 256, 0, stream>>>(logits, rmax, rsum, acc, s == 0 ? 0 : 1);
  }
}
